// BlockRasterizer_17514876634265
// MI455X (gfx1250) — matrix operand order repaired and checked by reading, not compiled, not run on hardware
//
#include <hip/hip_runtime.h>
#include <hip/hip_bf16.h>
#include <math.h>

// ---------------- problem constants (from reference) ----------------
#define R_    256
#define NB_   64
#define K_    8
#define S_    256
#define H_    128
#define FEAT_ 32
#define EMB_  16
#define NAPP_ 100
#define CIN_  52          // padded to 64 for WMMA
#define STEP_ 0.5f
#define FINF  __builtin_inff()

typedef __attribute__((ext_vector_type(16))) _Float16 v16h;
typedef __attribute__((ext_vector_type(8)))  _Float16 v8h;
typedef __attribute__((ext_vector_type(8)))  float    v8f;
typedef __attribute__((ext_vector_type(2)))  float    v2f;

// ---------------- LDS layout for the MLP kernel (bytes) ----------------
// h1    [256][128] f16 : 65536  (aliased later: cin [256][64] at 0, hc [256][64] at 32768)
// h2    [256][128] f16 : 65536
// W2^T  [128][128] f16 : 32768  (w2t[o][i])
// Bh^T  [ 48][128] f16 : 12288  (bh[n][i] = [Wf | Wd | 0]^T : feat cols 0..31, sigma col 32)
// Wc1^T [ 64][ 64] f16 :  8192  (wc1t[o][c], rows c=52..63 zero-padded)
// Wc2^T [ 16][ 64] f16 :  2048  (wc2t[n][c], cols n=3..15 zero-padded)
// small params (f32)   :  2912
#define OFF_H1    0
#define OFF_H2    65536
#define OFF_W2T   131072
#define OFF_BH    163840
#define OFF_WC1T  176128
#define OFF_WC2T  184320
#define OFF_SM    186368
#define SMEM_BYTES 189312

// ---------------- fragment load helpers (wave32 WMMA layouts, ISA 7.12.2) --------
// A (16x32 f16): lane row = m0 + (lane&15); halves [k0+hi*8 .. +7] and [k0+16+hi*8 .. +7]
static __device__ inline v16h ld_fragA(const _Float16* p /*16B aligned*/) {
  v8h lo = *(const v8h*)p;
  v8h hi = *(const v8h*)(p + 16);
  v16h r;
#pragma unroll
  for (int e = 0; e < 8; ++e) { r[e] = lo[e]; r[e + 8] = hi[e]; }
  return r;
}
// B (32x16 f16): lane col = n0 + (lane&15); halves [k0+hi*8 .. +7] and [k0+16+hi*8 .. +7]
static __device__ inline v16h ld_fragB(const _Float16* p /*16B aligned*/) {
  v8h lo = *(const v8h*)p;
  v8h hi = *(const v8h*)(p + 16);
  v16h r;
#pragma unroll
  for (int e = 0; e < 8; ++e) { r[e] = lo[e]; r[e + 8] = hi[e]; }
  return r;
}

// =====================================================================
// Kernel A: per-ray block selection (ray-sphere hit, nearest-K, stable)
// =====================================================================
__global__ __launch_bounds__(256)
void k_select(const float* __restrict__ o3, const float* __restrict__ d3,
              const float* __restrict__ bc, const float* __restrict__ br,
              int* __restrict__ selIdx, int* __restrict__ selValid)
{
  int r = blockIdx.x * blockDim.x + threadIdx.x;
  if (r >= R_) return;
  float ox = o3[r*3+0], oy = o3[r*3+1], oz = o3[r*3+2];
  float dx = d3[r*3+0], dy = d3[r*3+1], dz = d3[r*3+2];
  float a  = dx*dx + dy*dy + dz*dz;
  float dist[NB_];
#pragma unroll 4
  for (int b = 0; b < NB_; ++b) {
    float cx = bc[b*3+0], cy = bc[b*3+1], cz = bc[b*3+2];
    float ocx = ox - cx, ocy = oy - cy, ocz = oz - cz;
    float bq  = 2.f * (ocx*dx + ocy*dy + ocz*dz);
    float cq  = ocx*ocx + ocy*ocy + ocz*ocz - br[b]*br[b];
    float disc = bq*bq - 4.f*a*cq;
    float sq = (disc > 0.f) ? sqrtf(disc) : ((disc >= 0.f) ? 1.f : 0.f); // replicate ref exactly
    float t1 = (-bq - sq) / (2.f*a);
    float t2 = (-bq + sq) / (2.f*a);
    float th = (t1 > 0.f) ? t1 : t2;
    bool  valid = (disc >= 0.f) && (th > 0.f);
    float hx = ox + th*dx - cx, hy = oy + th*dy - cy, hz = oz + th*dz - cz;
    float ds = sqrtf(hx*hx + hy*hy + hz*hz);
    dist[b] = valid ? ds : FINF;
  }
  // top_k(-dist, 8): ascending distance, stable (lowest index wins ties)
  for (int j = 0; j < K_; ++j) {
    float best = FINF; int bi = 0;
    for (int b = 0; b < NB_; ++b)
      if (dist[b] < best) { best = dist[b]; bi = b; }
    selIdx[r*K_ + j]   = bi;
    selValid[r*K_ + j] = (best < 1e30f) ? 1 : 0;
    dist[bi] = FINF;
  }
}

// =====================================================================
// Kernel B: per-(ray, selected block) MLP over all 256 samples. WMMA everywhere.
// grid = R_*K_ = 2048 workgroups x 256 threads (8 waves)
// =====================================================================
__global__ __launch_bounds__(256, 1)
void k_mlp(const float* __restrict__ o3, const float* __restrict__ d3,
           const int*   __restrict__ appIds, const float* __restrict__ expo,
           const float* __restrict__ nearv,
           const float* __restrict__ W1, const float* __restrict__ b1,
           const float* __restrict__ W2, const float* __restrict__ b2,
           const float* __restrict__ Wd, const float* __restrict__ bdv,
           const float* __restrict__ Wf,
           const float* __restrict__ Wc1, const float* __restrict__ bc1,
           const float* __restrict__ Wc2, const float* __restrict__ bc2,
           const float* __restrict__ appEmb,
           const int* __restrict__ selIdx, const int* __restrict__ selValid,
           float* __restrict__ sigmaOut, float* __restrict__ colorOut)
{
  extern __shared__ char smem[];
  const int tid = threadIdx.x;
  const int r   = blockIdx.x >> 3;
  const int kk  = blockIdx.x & 7;
  if (!selValid[r*K_ + kk]) return;          // uniform across workgroup
  const int blk = selIdx[r*K_ + kk];

  _Float16* h1   = (_Float16*)(smem + OFF_H1);
  _Float16* h2   = (_Float16*)(smem + OFF_H2);
  _Float16* w2t  = (_Float16*)(smem + OFF_W2T);
  _Float16* bh   = (_Float16*)(smem + OFF_BH);
  _Float16* wc1t = (_Float16*)(smem + OFF_WC1T);
  _Float16* wc2t = (_Float16*)(smem + OFF_WC2T);
  _Float16* cin  = h1;                         // alias (h1 dead after layer 2)
  _Float16* hc   = (_Float16*)(smem + 32768);  // alias: upper half of h1 region
  float* sW1  = (float*)(smem + OFF_SM); // 384
  float* sB1  = sW1  + 384;              // 128
  float* sB2  = sB1  + 128;              // 128
  float* sBc1 = sB2  + 128;              // 64
  float* sBc2 = sBc1 + 64;               // 4
  float* sEmb = sBc2 + 4;                // 16
  float* sBd  = sEmb + 16;               // 4

  const float* gW2  = W2  + (size_t)blk * H_ * H_;
  const float* gWf  = Wf  + (size_t)blk * H_ * FEAT_;
  const float* gWd  = Wd  + (size_t)blk * H_;
  const float* gWc1 = Wc1 + (size_t)blk * CIN_ * 64;
  const float* gWc2 = Wc2 + (size_t)blk * 64 * 3;
  __builtin_prefetch(gW2,  0, 1);   // -> global_prefetch_b8
  __builtin_prefetch(gWc1, 0, 1);

  // ---- stage weights into LDS as f16, transposed for B-fragments
  for (int t = tid; t < H_*H_; t += 256) {          // w2t[o][i] = W2[i][o]
    int o = t >> 7, i = t & 127;
    w2t[t] = (_Float16)gW2[i*H_ + o];
  }
  for (int t = tid; t < 48*H_; t += 256) {          // bh[n][i] = [Wf | Wd | 0]
    int n = t >> 7, i = t & 127;
    float v = (n < FEAT_) ? gWf[i*FEAT_ + n] : ((n == FEAT_) ? gWd[i] : 0.f);
    bh[t] = (_Float16)v;
  }
  for (int t = tid; t < 64*64; t += 256) {          // wc1t[o][c] = Wc1[c][o], pad c>=52
    int o = t >> 6, c = t & 63;
    wc1t[t] = (c < CIN_) ? (_Float16)gWc1[c*64 + o] : (_Float16)0.f;
  }
  for (int t = tid; t < 16*64; t += 256) {          // wc2t[n][c] = Wc2[c][n], pad n>=3
    int n = t >> 6, c = t & 63;
    wc2t[t] = (n < 3) ? (_Float16)gWc2[c*3 + n] : (_Float16)0.f;
  }
  if (tid < 384) sW1[tid]  = W1[blk*384 + tid];
  if (tid < 128) { sB1[tid] = b1[blk*H_ + tid]; sB2[tid] = b2[blk*H_ + tid]; }
  if (tid < 64)  sBc1[tid] = bc1[blk*64 + tid];
  if (tid < 3)   sBc2[tid] = bc2[blk*3 + tid];
  if (tid < EMB_) sEmb[tid] = appEmb[((size_t)blk*NAPP_ + appIds[r])*EMB_ + tid];
  if (tid == 0)  sBd[0] = bdv[blk];

  const float ox = o3[r*3], oy = o3[r*3+1], oz = o3[r*3+2];
  const float dx = d3[r*3], dy = d3[r*3+1], dz = d3[r*3+2];
  const float tnear = nearv[r];
  __syncthreads();

  const int lane   = tid & 31;
  const int wv     = tid >> 5;      // wave id 0..7
  const int laneLo = lane & 15;
  const int laneHi = lane >> 4;

  // ---- h1 = relu([pos,1] @ [W1;b1]) via V_WMMA_F32_16X16X4_F32 (bias folded in K=4)
  // A 16x4 f32: lanes 0-15 hold K=(0,1)=(px,py); lanes 16-31 hold K=(2,3)=(pz,1)
  // B 4x16 f32: lanes 0-15 hold K=(0,1); lanes 16-31 hold K=(2,3); col = n
  {
    const int n = wv*16 + laneLo;
    v2f bf;
    bf[0] = laneHi ? sW1[256 + n] : sW1[n];
    bf[1] = laneHi ? sB1[n]       : sW1[128 + n];
    for (int mt = 0; mt < 16; ++mt) {
      const float srow = (float)(mt*16 + laneLo);
      const float tpos = tnear + STEP_ * srow;
      v2f af;
      af[0] = laneHi ? (oz + tpos*dz) : (ox + tpos*dx);
      af[1] = laneHi ? 1.0f           : (oy + tpos*dy);
      v8f acc = {0.f,0.f,0.f,0.f,0.f,0.f,0.f,0.f};
      acc = __builtin_amdgcn_wmma_f32_16x16x4_f32(false, af, false, bf,
                                                  (short)0, acc, false, false);
      _Float16* out = h1 + (mt*16 + laneHi*8)*H_ + n;
#pragma unroll
      for (int e = 0; e < 8; ++e) out[e*H_] = (_Float16)fmaxf(acc[e], 0.f);
    }
  }
  __syncthreads();

  // ---- layer 2: h2 = relu(h1 @ W2 + b2)   M=256 N=128 K=128
  {
    v16h bf[4];
    const _Float16* bp = w2t + (wv*16 + laneLo)*H_ + laneHi*8;
#pragma unroll
    for (int ks = 0; ks < 4; ++ks) bf[ks] = ld_fragB(bp + ks*32);
    const float bias = sB2[wv*16 + laneLo];
    for (int mt = 0; mt < 16; ++mt) {
      v8f acc;
#pragma unroll
      for (int e = 0; e < 8; ++e) acc[e] = bias;
#pragma unroll
      for (int ks = 0; ks < 4; ++ks) {
        v16h af = ld_fragA(h1 + (mt*16 + laneLo)*H_ + ks*32 + laneHi*8);
        acc = __builtin_amdgcn_wmma_f32_16x16x32_f16(false, af, false, bf[ks],
                                                     (short)0, acc, false, false);
      }
      _Float16* out = h2 + (mt*16 + laneHi*8)*H_ + wv*16 + laneLo;
#pragma unroll
      for (int e = 0; e < 8; ++e) out[e*H_] = (_Float16)fmaxf(acc[e], 0.f);
    }
  }
  __syncthreads();

  // ---- cin constant columns (per-sample, cheap VALU)
  {
    const int s = tid;
    _Float16* crow = cin + s*64;           // cols 32..63: [d(3), emb(16), exp(1), pad(12)]
    crow[32] = (_Float16)dx; crow[33] = (_Float16)dy; crow[34] = (_Float16)dz;
#pragma unroll
    for (int e = 0; e < EMB_; ++e) crow[35 + e] = (_Float16)sEmb[e];
    crow[51] = (_Float16)expo[r];
#pragma unroll
    for (int e = 52; e < 64; ++e) crow[e] = (_Float16)0.f;
  }
  // ---- fused head: [feat | sigma] = h2 @ [Wf | Wd]   M=256 N=48 K=128
  // N-tiles 0,1 -> feat into cin cols 0..31 ; N-tile 2 col 32 -> softplus(sigma) to global
  {
    const float bd0 = sBd[0];
    for (int j = 0; j < 6; ++j) {
      const int t  = wv + 8*j;          // 48 tiles over 8 waves
      const int nt = t % 3;
      const int mt = t / 3;
      v16h bf[4];
      const _Float16* bp = bh + (nt*16 + laneLo)*H_ + laneHi*8;
#pragma unroll
      for (int ks = 0; ks < 4; ++ks) bf[ks] = ld_fragB(bp + ks*32);
      v8f acc = {0.f,0.f,0.f,0.f,0.f,0.f,0.f,0.f};
#pragma unroll
      for (int ks = 0; ks < 4; ++ks) {
        v16h af = ld_fragA(h2 + (mt*16 + laneLo)*H_ + ks*32 + laneHi*8);
        acc = __builtin_amdgcn_wmma_f32_16x16x32_f16(false, af, false, bf[ks],
                                                     (short)0, acc, false, false);
      }
      if (nt < 2) {
        _Float16* out = cin + (mt*16 + laneHi*8)*64 + nt*16 + laneLo;
#pragma unroll
        for (int e = 0; e < 8; ++e) out[e*64] = (_Float16)acc[e];
      } else if (laneLo == 0) {        // N=32 column: density head
#pragma unroll
        for (int e = 0; e < 8; ++e) {
          const int s = mt*16 + laneHi*8 + e;
          const float x = acc[e] + bd0;
          const float sp = fmaxf(x, 0.f) + log1pf(expf(-fabsf(x)));  // stable softplus
          sigmaOut[((size_t)r*S_ + s)*K_ + kk] = sp;
        }
      }
    }
  }
  __syncthreads();

  // ---- hc = relu(cin @ Wc1 + bc1)   M=256 N=64 K=64 (zero-padded K)
  {
    const int nt = wv & 3;
    const int mb = wv >> 2;
    v16h bf[2];
    const _Float16* bp = wc1t + (nt*16 + laneLo)*64 + laneHi*8;
#pragma unroll
    for (int ks = 0; ks < 2; ++ks) bf[ks] = ld_fragB(bp + ks*32);
    const float bias = sBc1[nt*16 + laneLo];
    for (int mi = 0; mi < 8; ++mi) {
      const int mt = mb + mi*2;
      v8f acc;
#pragma unroll
      for (int e = 0; e < 8; ++e) acc[e] = bias;
#pragma unroll
      for (int ks = 0; ks < 2; ++ks) {
        v16h af = ld_fragA(cin + (mt*16 + laneLo)*64 + ks*32 + laneHi*8);
        acc = __builtin_amdgcn_wmma_f32_16x16x32_f16(false, af, false, bf[ks],
                                                     (short)0, acc, false, false);
      }
      _Float16* out = hc + (mt*16 + laneHi*8)*64 + nt*16 + laneLo;
#pragma unroll
      for (int e = 0; e < 8; ++e) out[e*64] = (_Float16)fmaxf(acc[e], 0.f);
    }
  }
  __syncthreads();

  // ---- color = sigmoid(hc @ Wc2 + bc2) via WMMA (N=16, cols 0..2 used)
  {
    v16h bf[2];
    const _Float16* bp = wc2t + laneLo*64 + laneHi*8;
#pragma unroll
    for (int ks = 0; ks < 2; ++ks) bf[ks] = ld_fragB(bp + ks*32);
    const float bias = (laneLo < 3) ? sBc2[laneLo] : 0.f;
    for (int mi = 0; mi < 2; ++mi) {
      const int mt = wv + 8*mi;
      v8f acc = {0.f,0.f,0.f,0.f,0.f,0.f,0.f,0.f};
#pragma unroll
      for (int ks = 0; ks < 2; ++ks) {
        v16h af = ld_fragA(hc + (mt*16 + laneLo)*64 + ks*32 + laneHi*8);
        acc = __builtin_amdgcn_wmma_f32_16x16x32_f16(false, af, false, bf[ks],
                                                     (short)0, acc, false, false);
      }
      if (laneLo < 3) {
#pragma unroll
        for (int e = 0; e < 8; ++e) {
          const int s = mt*16 + laneHi*8 + e;
          const float c = acc[e] + bias;
          colorOut[(((size_t)r*S_ + s)*K_ + kk)*3 + laneLo] = 1.f/(1.f + expf(-c));
        }
      }
    }
  }
}

// =====================================================================
// Kernel C: per-ray blend across K + sequential compositing over S
// =====================================================================
__global__ __launch_bounds__(256)
void k_composite(const float* __restrict__ o3, const float* __restrict__ d3,
                 const float* __restrict__ bc,
                 const float* __restrict__ nearv, const float* __restrict__ farv,
                 const int* __restrict__ selIdx, const int* __restrict__ selValid,
                 const float* __restrict__ sigmaW, const float* __restrict__ colorW,
                 float* __restrict__ outRgb, float* __restrict__ outDepth,
                 float* __restrict__ outAcc)
{
  __shared__ float sA[S_], sC0[S_], sC1[S_], sC2[S_], sTt[S_];
  __shared__ float sCx[K_], sCy[K_], sCz[K_];
  __shared__ int   sVal[K_];
  const int r = blockIdx.x;
  const int s = threadIdx.x;
  if (s < K_) {
    int bi = selIdx[r*K_ + s];
    sVal[s] = selValid[r*K_ + s];
    sCx[s] = bc[bi*3]; sCy[s] = bc[bi*3+1]; sCz[s] = bc[bi*3+2];
  }
  __syncthreads();

  const float ox = o3[r*3], oy = o3[r*3+1], oz = o3[r*3+2];
  const float dx = d3[r*3], dy = d3[r*3+1], dz = d3[r*3+2];
  const float tpos = nearv[r] + STEP_ * (float)s;
  const float px = ox + tpos*dx, py = oy + tpos*dy, pz = oz + tpos*dz;

  float inv[K_]; float wsum = 0.f;
#pragma unroll
  for (int k = 0; k < K_; ++k) {
    float ddx = px - sCx[k], ddy = py - sCy[k], ddz = pz - sCz[k];
    float dist = sqrtf(ddx*ddx + ddy*ddy + ddz*ddz);
    inv[k] = sVal[k] ? (1.f / (dist + 1e-6f)) : 0.f;
    wsum += inv[k];
  }
  float td = 0.f, tc0 = 0.f, tc1 = 0.f, tc2 = 0.f, tw = 0.f;
#pragma unroll
  for (int k = 0; k < K_; ++k) {
    float w  = (wsum > 0.f) ? (inv[k] / fmaxf(wsum, 1e-12f)) : 0.f;
    float wvv = ((w >= 0.01f) && sVal[k]) ? w : 0.f;
    size_t base = ((size_t)r*S_ + s)*K_ + k;
    td  += sigmaW[base] * wvv;
    tc0 += colorW[base*3+0] * wvv;
    tc1 += colorW[base*3+1] * wvv;
    tc2 += colorW[base*3+2] * wvv;
    tw  += wvv;
  }
  float itw = (tw > 0.f) ? (1.f / fmaxf(tw, 1e-12f)) : 1.f;
  float alpha = 1.f - expf(-td * STEP_);
  alpha = fminf(fmaxf(alpha, 0.f), 1.f);
  sA[s] = alpha; sC0[s] = tc0*itw; sC1[s] = tc1*itw; sC2[s] = tc2*itw; sTt[s] = tpos;
  __syncthreads();

  if (s == 0) {
    float T = 1.f, acc = 0.f, r0 = 0.f, r1 = 0.f, r2 = 0.f, dep = 0.f;
    const float fr = farv[r];
    for (int i = 0; i < S_; ++i) {
      bool alive = (sTt[i] < fr) && (T > 1e-4f) && (acc <= 0.99f);
      float wgt = alive ? T * sA[i] : 0.f;
      r0 += wgt * sC0[i]; r1 += wgt * sC1[i]; r2 += wgt * sC2[i];
      acc += wgt; dep += wgt * sTt[i];
      T *= alive ? (1.f - sA[i]) : 1.f;
    }
    outRgb[r*3+0] = r0; outRgb[r*3+1] = r1; outRgb[r*3+2] = r2;
    outDepth[r] = dep / fmaxf(acc, 1e-6f);
    outAcc[r]   = acc;
  }
}

// =====================================================================
extern "C" void kernel_launch(void* const* d_in, const int* in_sizes, int n_in,
                              void* d_out, int out_size, void* d_ws, size_t ws_size,
                              hipStream_t stream) {
  const float* rayO  = (const float*)d_in[0];
  const float* rayD  = (const float*)d_in[1];
  const float* bcen  = (const float*)d_in[2];
  const float* brad  = (const float*)d_in[3];
  const int*   appId = (const int*)  d_in[4];
  const float* expo  = (const float*)d_in[5];
  const float* nearv = (const float*)d_in[6];
  const float* farv  = (const float*)d_in[7];
  const float* W1    = (const float*)d_in[8];
  const float* b1    = (const float*)d_in[9];
  const float* W2    = (const float*)d_in[10];
  const float* b2    = (const float*)d_in[11];
  const float* Wd    = (const float*)d_in[12];
  const float* bdv   = (const float*)d_in[13];
  const float* Wf    = (const float*)d_in[14];
  const float* Wc1   = (const float*)d_in[15];
  const float* bc1   = (const float*)d_in[16];
  const float* Wc2   = (const float*)d_in[17];
  const float* bc2   = (const float*)d_in[18];
  const float* aemb  = (const float*)d_in[19];

  int*   selIdx   = (int*)d_ws;
  int*   selValid = selIdx + R_*K_;
  float* sigmaW   = (float*)((char*)d_ws + 16384);
  float* colorW   = sigmaW + (size_t)R_*S_*K_;
  float* outRgb   = (float*)d_out;
  float* outDepth = outRgb + R_*3;
  float* outAcc   = outDepth + R_;

  k_select<<<1, 256, 0, stream>>>(rayO, rayD, bcen, brad, selIdx, selValid);

  (void)hipFuncSetAttribute(reinterpret_cast<const void*>(k_mlp),
                            hipFuncAttributeMaxDynamicSharedMemorySize, SMEM_BYTES);
  k_mlp<<<R_*K_, 256, SMEM_BYTES, stream>>>(rayO, rayD, appId, expo, nearv,
                                            W1, b1, W2, b2, Wd, bdv, Wf,
                                            Wc1, bc1, Wc2, bc2, aemb,
                                            selIdx, selValid, sigmaW, colorW);

  k_composite<<<R_, 256, 0, stream>>>(rayO, rayD, bcen, nearv, farv,
                                      selIdx, selValid, sigmaW, colorW,
                                      outRgb, outDepth, outAcc);
}
